// GINGATRegressor_30820685316247
// MI455X (gfx1250) — matrix operand order repaired and checked by reading, not compiled, not run on hardware
//
#include <hip/hip_runtime.h>
#include <hip/hip_bf16.h>

typedef __attribute__((ext_vector_type(16))) _Float16 v16h;
typedef __attribute__((ext_vector_type(8)))  float    v8f;

#define NN   30000
#define EE   480000
#define GG   64
#define DD   512
#define HH   4
#define CC   128
#define HID  128
#define BN_EPS 1e-5f

// ---------------------------------------------------------------------------
// WMMA GEMM: C[M,Nc] = A[M,K](f16) * Bt[Nc,K]^T(f16) (+bias)(+relu)
// One 16x16 tile per wave. K multiple of 32, M,Nc multiples of 16.
// Fragment layouts per CDNA5 ISA 7.12.2:
//   A 16x32 f16: lane<16 row=lane K={k..k+7, k+16..k+23}; lane>=16 +8
//   B 32x16 f16: lane&15=col; K as for A: lane<16 K={k..k+7, k+16..k+23}; lane>=16 +8
//   C/D 16x16 f32: vgpr r -> row = r + 8*(lane>=16), col = lane&15
// ---------------------------------------------------------------------------
__global__ __launch_bounds__(256)
void wmma_gemm_kernel(const _Float16* __restrict__ A, const _Float16* __restrict__ Bt,
                      const float* __restrict__ bias, float* __restrict__ C,
                      int M, int K, int Nc, int ldc, int col_off, int relu) {
  int lane = threadIdx.x & 31;
  int wave = threadIdx.x >> 5;
  int tile = blockIdx.x * 8 + wave;
  int tilesN = Nc >> 4;
  int total = (M >> 4) * tilesN;
  if (tile >= total) return;                 // wave-uniform: EXEC all-1 in WMMA
  int tm = tile / tilesN;
  int tn = tile - tm * tilesN;
  int hl  = lane & 15;
  int sel = lane >> 4;

  const _Float16* ap = A  + (size_t)(tm * 16 + hl) * K;
  const _Float16* bp = Bt + (size_t)(tn * 16 + hl) * K;

  v8f acc = {};
  for (int k = 0; k < K; k += 32) {
    v16h a, b;
    const _Float16* a0 = ap + k + sel * 8;       // 8 contiguous halfs (16B)
    const _Float16* a1 = ap + k + 16 + sel * 8;  // 8 contiguous halfs (16B)
#pragma unroll
    for (int i = 0; i < 8; ++i) { a[i] = a0[i]; a[8 + i] = a1[i]; }
    const _Float16* b0 = bp + k + sel * 8;       // 8 contiguous halfs (16B)
    const _Float16* b1 = bp + k + 16 + sel * 8;  // 8 contiguous halfs (16B)
#pragma unroll
    for (int i = 0; i < 8; ++i) { b[i] = b0[i]; b[8 + i] = b1[i]; }
    acc = __builtin_amdgcn_wmma_f32_16x16x32_f16(false, a, false, b,
                                                 (short)0, acc, false, false);
  }

  int col = tn * 16 + hl;
  float bv = bias ? bias[col] : 0.0f;
#pragma unroll
  for (int r = 0; r < 8; ++r) {
    int row = tm * 16 + r + 8 * sel;
    float v = acc[r] + bv;
    if (relu) v = v > 0.0f ? v : 0.0f;
    C[(size_t)row * ldc + col_off + col] = v;
  }
}

// ---------------------------------------------------------------------------
// Small helper kernels
// ---------------------------------------------------------------------------
__global__ void copy_f32_kernel(const float* __restrict__ x, float* __restrict__ y, unsigned n) {
  unsigned t = blockIdx.x * blockDim.x + threadIdx.x;
  if (t < n) y[t] = x[t];
}
__global__ void set_f32_kernel(float* __restrict__ y, float v, unsigned n) {
  unsigned t = blockIdx.x * blockDim.x + threadIdx.x;
  if (t < n) y[t] = v;
}
__global__ void f32_to_f16_kernel(const float* __restrict__ x, _Float16* __restrict__ y, unsigned n) {
  unsigned t = blockIdx.x * blockDim.x + threadIdx.x;
  if (t < n) y[t] = (_Float16)x[t];
}
// pad [N,20] -> [N,32] f16 (zeros in cols 20..31)
__global__ void pad20_f16_kernel(const float* __restrict__ x, _Float16* __restrict__ y, int n) {
  unsigned t = blockIdx.x * blockDim.x + threadIdx.x;
  if (t >= (unsigned)n * 32u) return;
  int row = t >> 5, c = t & 31;
  y[t] = (c < 20) ? (_Float16)x[row * 20 + c] : (_Float16)0.0f;
}
// Wt[Nc,Kp] = transpose(W[K,Nc]) converted to f16, zero-padded K->Kp
__global__ void wtransp_f16_kernel(const float* __restrict__ W, _Float16* __restrict__ Wt,
                                   int K, int Nc, int Kp) {
  unsigned t = blockIdx.x * blockDim.x + threadIdx.x;
  if (t >= (unsigned)Nc * (unsigned)Kp) return;
  int n = t / Kp, k = t - n * Kp;
  Wt[t] = (k < K) ? (_Float16)W[(size_t)k * Nc + n] : (_Float16)0.0f;
}

// edge aggregation: agg[dst] += x[src], one element per thread
__global__ void edge_agg_kernel(const int* __restrict__ ei, const float* __restrict__ x,
                                float* __restrict__ agg, int E, int D) {
  unsigned t = blockIdx.x * blockDim.x + threadIdx.x;
  if (t >= (unsigned)E * (unsigned)D) return;
  int e = t / D, c = t - e * D;
  int s = ei[e], d = ei[E + e];
  atomicAdd(&agg[(size_t)d * D + c], x[(size_t)s * D + c]);
}

// batch-norm: column stats (mean, rsqrt(var+eps)) over `rows`
__global__ __launch_bounds__(256)
void bn_stats_kernel(const float* __restrict__ X, float* __restrict__ stats, int rows, int cols) {
  int col = blockIdx.x;
  float s = 0.f, sq = 0.f;
  for (int r = threadIdx.x; r < rows; r += 256) {
    float v = X[(size_t)r * cols + col];
    s += v; sq += v * v;
  }
  __shared__ float sh[512];
  sh[threadIdx.x] = s; sh[256 + threadIdx.x] = sq;
  __syncthreads();
  for (int st = 128; st > 0; st >>= 1) {
    if ((int)threadIdx.x < st) {
      sh[threadIdx.x] += sh[threadIdx.x + st];
      sh[256 + threadIdx.x] += sh[256 + threadIdx.x + st];
    }
    __syncthreads();
  }
  if (threadIdx.x == 0) {
    float m = sh[0] / rows;
    float var = sh[256] / rows - m * m;
    stats[col] = m;
    stats[cols + col] = rsqrtf(var + BN_EPS);
  }
}
__global__ void bn_apply_kernel(const float* __restrict__ X, const float* __restrict__ stats,
                                const float* __restrict__ g, const float* __restrict__ b,
                                float* __restrict__ Y, int rows, int cols) {
  unsigned t = blockIdx.x * blockDim.x + threadIdx.x;
  if (t >= (unsigned)rows * (unsigned)cols) return;
  int c = t % cols;
  float v = (X[t] - stats[c]) * stats[cols + c] * g[c] + b[c];
  Y[t] = v > 0.0f ? v : 0.0f;    // all BN uses in this model are followed by ReLU
}

// ------------------------- GAT -------------------------
__device__ __forceinline__ unsigned fenc(float f) {
  unsigned u = __float_as_uint(f);
  return (u & 0x80000000u) ? ~u : (u | 0x80000000u);
}
__device__ __forceinline__ float fdec(unsigned u) {
  return (u & 0x80000000u) ? __uint_as_float(u & 0x7fffffffu) : __uint_as_float(~u);
}

__global__ void gat_scores_kernel(const float* __restrict__ hh, const float* __restrict__ as,
                                  const float* __restrict__ ad, float* __restrict__ a_s,
                                  float* __restrict__ a_d, int n) {
  unsigned t = blockIdx.x * blockDim.x + threadIdx.x;
  if (t >= (unsigned)n * HH) return;
  int node = t >> 2, h = t & 3;
  const float* hp = hh + (size_t)node * DD + h * CC;
  const float* pa = as + h * CC;
  const float* pb = ad + h * CC;
  float s = 0.f, d = 0.f;
#pragma unroll 4
  for (int c = 0; c < CC; ++c) { float v = hp[c]; s += v * pa[c]; d += v * pb[c]; }
  a_s[t] = s; a_d[t] = d;
}
__global__ void gat_init_kernel(unsigned* __restrict__ mx, float* __restrict__ den, unsigned n) {
  unsigned t = blockIdx.x * blockDim.x + threadIdx.x;
  if (t < n) { mx[t] = fenc(-3.0e38f); den[t] = 0.0f; }
}
// pass 1: e = leakyrelu(a_s[src]+a_d[dst], 0.2); esc=e; segment max
__global__ void gat_edge1_kernel(const int* __restrict__ ei, const float* __restrict__ a_s,
                                 const float* __restrict__ a_d, float* __restrict__ esc,
                                 unsigned* __restrict__ mx, int E, int n) {
  unsigned t = blockIdx.x * blockDim.x + threadIdx.x;
  if (t >= (unsigned)(E + n) * HH) return;
  int e2 = t >> 2, h = t & 3;
  int s, d;
  if (e2 < E) { s = ei[e2]; d = ei[E + e2]; } else { s = d = e2 - E; }
  float ev = a_s[s * HH + h] + a_d[d * HH + h];
  ev = ev > 0.0f ? ev : 0.2f * ev;
  esc[t] = ev;
  atomicMax(&mx[d * HH + h], fenc(ev));
}
// pass 2: ex = exp(e - mx[dst]); esc=ex; den[dst]+=ex
__global__ void gat_edge2_kernel(const int* __restrict__ ei, float* __restrict__ esc,
                                 const unsigned* __restrict__ mx, float* __restrict__ den,
                                 int E, int n) {
  unsigned t = blockIdx.x * blockDim.x + threadIdx.x;
  if (t >= (unsigned)(E + n) * HH) return;
  int e2 = t >> 2, h = t & 3;
  int d = (e2 < E) ? ei[E + e2] : e2 - E;
  float ex = expf(esc[t] - fdec(mx[d * HH + h]));
  esc[t] = ex;
  atomicAdd(&den[d * HH + h], ex);
}
__global__ void gat_bias_bcast_kernel(const float* __restrict__ bias, float* __restrict__ out,
                                      unsigned n) {
  unsigned t = blockIdx.x * blockDim.x + threadIdx.x;
  if (t < n) out[t] = bias[t & (DD - 1)];
}
// pass 3: out[dst] += alpha * hh[src]; 4 channels (one head) per thread
__global__ void gat_edge3_kernel(const int* __restrict__ ei, const float* __restrict__ hh,
                                 const float* __restrict__ esc, const float* __restrict__ den,
                                 float* __restrict__ out, int E, int n) {
  unsigned t = blockIdx.x * blockDim.x + threadIdx.x;
  if (t >= (unsigned)(E + n) * (DD / 4)) return;
  int e2 = t >> 7;                // 128 quads per edge
  int c4 = (t & 127) * 4;
  int h = c4 >> 7;
  int s, d;
  if (e2 < E) { s = ei[e2]; d = ei[E + e2]; } else { s = d = e2 - E; }
  float alpha = esc[(size_t)e2 * HH + h] / den[d * HH + h];
  const float4 v = *(const float4*)(hh + (size_t)s * DD + c4);
  float* o = out + (size_t)d * DD + c4;
  atomicAdd(o + 0, v.x * alpha);
  atomicAdd(o + 1, v.y * alpha);
  atomicAdd(o + 2, v.z * alpha);
  atomicAdd(o + 3, v.w * alpha);
}

__global__ void pool_kernel(const int* __restrict__ batch, const float* __restrict__ x,
                            float* __restrict__ g, int n) {
  unsigned t = blockIdx.x * blockDim.x + threadIdx.x;
  if (t >= (unsigned)n * DD) return;
  int node = t >> 9, f = t & (DD - 1);
  atomicAdd(&g[(size_t)batch[node] * DD + f], x[t]);
}

__global__ void reg_final_kernel(const float* __restrict__ rh, const float* __restrict__ W2,
                                 const float* __restrict__ b2, float* __restrict__ out) {
  int g = blockIdx.x * blockDim.x + threadIdx.x;
  if (g >= GG) return;
  float s = b2[0];
#pragma unroll 4
  for (int k = 0; k < HID; ++k) s += rh[g * HID + k] * W2[k];
  out[g] = s;
}

// ---------------------------------------------------------------------------
// Host orchestration
// ---------------------------------------------------------------------------
static inline unsigned cdiv_u(unsigned a, unsigned b) { return (a + b - 1) / b; }

static void gemm(const _Float16* A, const _Float16* Bt, const float* bias, float* C,
                 int M, int K, int Nc, int ldc, int col_off, int relu, hipStream_t st) {
  int tiles = (M / 16) * (Nc / 16);
  wmma_gemm_kernel<<<dim3((tiles + 7) / 8), dim3(256), 0, st>>>(A, Bt, bias, C, M, K, Nc,
                                                                ldc, col_off, relu);
}
static void conv16(const float* x, _Float16* y, unsigned n, hipStream_t st) {
  f32_to_f16_kernel<<<cdiv_u(n, 256), 256, 0, st>>>(x, y, n);
}
static void wtr(const float* W, _Float16* Wt, int K, int Nc, int Kp, hipStream_t st) {
  unsigned n = (unsigned)Nc * (unsigned)Kp;
  wtransp_f16_kernel<<<cdiv_u(n, 256), 256, 0, st>>>(W, Wt, K, Nc, Kp);
}

static void run_branch(const float* x, const int* ei, const int* batch,
                       const float* g1W1, const float* g1b1, const float* g1W2, const float* g1b2,
                       const float* bn1g, const float* bn1b,
                       const float* g2W1, const float* g2b1, const float* g2W2, const float* g2b2,
                       const float* bn2g, const float* bn2b,
                       const float* gatW, const float* gatas, const float* gatad, const float* gatbias,
                       const float* fcW1, const float* fcb1, const float* fcbng, const float* fcbnb,
                       const float* fcW2, const float* fcb2,
                       float* buf0, float* buf1, float* buf2, _Float16* A16, _Float16* Wt16,
                       float* stats, float* a_s, float* a_d, unsigned* mx, float* den, float* esc,
                       float* gpool, float* gbuf, _Float16* g16, float* xcat, int col_off,
                       hipStream_t st) {
  const unsigned ND = (unsigned)NN * DD;
  const unsigned N20 = (unsigned)NN * 20u;

  // ---- GIN1: h = relu(bn(mlp(x + agg(x)))) ----
  copy_f32_kernel<<<cdiv_u(N20, 256), 256, 0, st>>>(x, buf1, N20);
  edge_agg_kernel<<<cdiv_u((unsigned)EE * 20u, 256), 256, 0, st>>>(ei, x, buf1, EE, 20);
  pad20_f16_kernel<<<cdiv_u((unsigned)NN * 32u, 256), 256, 0, st>>>(buf1, A16, NN);
  wtr(g1W1, Wt16, 20, DD, 32, st);
  gemm(A16, Wt16, g1b1, buf0, NN, 32, DD, DD, 0, 1, st);        // relu(x@W1+b1)
  conv16(buf0, A16, ND, st);
  wtr(g1W2, Wt16, DD, DD, DD, st);
  gemm(A16, Wt16, g1b2, buf1, NN, DD, DD, DD, 0, 0, st);        // @W2+b2
  bn_stats_kernel<<<DD, 256, 0, st>>>(buf1, stats, NN, DD);
  bn_apply_kernel<<<cdiv_u(ND, 256), 256, 0, st>>>(buf1, stats, bn1g, bn1b, buf0, NN, DD);

  // ---- GIN2 ----
  copy_f32_kernel<<<cdiv_u(ND, 256), 256, 0, st>>>(buf0, buf2, ND);
  edge_agg_kernel<<<cdiv_u((unsigned)EE * DD, 256), 256, 0, st>>>(ei, buf0, buf2, EE, DD);
  conv16(buf2, A16, ND, st);
  wtr(g2W1, Wt16, DD, DD, DD, st);
  gemm(A16, Wt16, g2b1, buf1, NN, DD, DD, DD, 0, 1, st);
  conv16(buf1, A16, ND, st);
  wtr(g2W2, Wt16, DD, DD, DD, st);
  gemm(A16, Wt16, g2b2, buf2, NN, DD, DD, DD, 0, 0, st);
  bn_stats_kernel<<<DD, 256, 0, st>>>(buf2, stats, NN, DD);
  bn_apply_kernel<<<cdiv_u(ND, 256), 256, 0, st>>>(buf2, stats, bn2g, bn2b, buf0, NN, DD);

  // ---- GAT ----
  conv16(buf0, A16, ND, st);
  wtr(gatW, Wt16, DD, DD, DD, st);
  gemm(A16, Wt16, nullptr, buf1, NN, DD, DD, DD, 0, 0, st);     // hh = h2 @ gat_W
  gat_scores_kernel<<<cdiv_u((unsigned)NN * HH, 256), 256, 0, st>>>(buf1, gatas, gatad,
                                                                    a_s, a_d, NN);
  gat_init_kernel<<<cdiv_u((unsigned)NN * HH, 256), 256, 0, st>>>(mx, den, (unsigned)NN * HH);
  unsigned eh = (unsigned)(EE + NN) * HH;
  gat_edge1_kernel<<<cdiv_u(eh, 256), 256, 0, st>>>(ei, a_s, a_d, esc, mx, EE, NN);
  gat_edge2_kernel<<<cdiv_u(eh, 256), 256, 0, st>>>(ei, esc, mx, den, EE, NN);
  gat_bias_bcast_kernel<<<cdiv_u(ND, 256), 256, 0, st>>>(gatbias, buf2, ND);
  gat_edge3_kernel<<<cdiv_u((unsigned)(EE + NN) * (DD / 4), 256), 256, 0, st>>>(
      ei, buf1, esc, den, buf2, EE, NN);

  // ---- global add pool + fc head ----
  set_f32_kernel<<<cdiv_u(GG * DD, 256), 256, 0, st>>>(gpool, 0.0f, GG * DD);
  pool_kernel<<<cdiv_u(ND, 256), 256, 0, st>>>(batch, buf2, gpool, NN);
  conv16(gpool, g16, GG * DD, st);
  wtr(fcW1, Wt16, DD, DD, DD, st);
  gemm(g16, Wt16, fcb1, gbuf, GG, DD, DD, DD, 0, 0, st);
  bn_stats_kernel<<<DD, 256, 0, st>>>(gbuf, stats, GG, DD);
  bn_apply_kernel<<<cdiv_u(GG * DD, 256), 256, 0, st>>>(gbuf, stats, fcbng, fcbnb, gpool, GG, DD);
  conv16(gpool, g16, GG * DD, st);
  wtr(fcW2, Wt16, DD, HID, DD, st);
  gemm(g16, Wt16, fcb2, xcat, GG, DD, HID, 2 * HID, col_off, 0, st);  // into concat slot
}

extern "C" void kernel_launch(void* const* d_in, const int* in_sizes, int n_in,
                              void* d_out, int out_size, void* d_ws, size_t ws_size,
                              hipStream_t stream) {
  const float* wl_x = (const float*)d_in[0];
  const float* mt_x = (const float*)d_in[1];
  const int* wl_ei = (const int*)d_in[2];
  const int* mt_ei = (const int*)d_in[3];
  const int* wl_b = (const int*)d_in[4];
  const int* mt_b = (const int*)d_in[5];
  const float* g1W1 = (const float*)d_in[6];  const float* g1b1 = (const float*)d_in[7];
  const float* g1W2 = (const float*)d_in[8];  const float* g1b2 = (const float*)d_in[9];
  const float* bn1g = (const float*)d_in[10]; const float* bn1b = (const float*)d_in[11];
  const float* g2W1 = (const float*)d_in[12]; const float* g2b1 = (const float*)d_in[13];
  const float* g2W2 = (const float*)d_in[14]; const float* g2b2 = (const float*)d_in[15];
  const float* bn2g = (const float*)d_in[16]; const float* bn2b = (const float*)d_in[17];
  const float* gatW = (const float*)d_in[18];
  const float* gatas = (const float*)d_in[19]; const float* gatad = (const float*)d_in[20];
  const float* gatbias = (const float*)d_in[21];
  const float* fcW1 = (const float*)d_in[22]; const float* fcb1 = (const float*)d_in[23];
  const float* fcbng = (const float*)d_in[24]; const float* fcbnb = (const float*)d_in[25];
  const float* fcW2 = (const float*)d_in[26]; const float* fcb2 = (const float*)d_in[27];
  const float* rW1 = (const float*)d_in[28]; const float* rb1 = (const float*)d_in[29];
  const float* rW2 = (const float*)d_in[30]; const float* rb2 = (const float*)d_in[31];

  // ---- workspace layout ----
  char* w = (char*)d_ws;
  auto alloc = [&](size_t bytes) -> void* {
    void* p = (void*)w;
    w += (bytes + 255) & ~(size_t)255;
    return p;
  };
  const size_t ND = (size_t)NN * DD;
  float*     buf0  = (float*)alloc(ND * 4);
  float*     buf1  = (float*)alloc(ND * 4);
  float*     buf2  = (float*)alloc(ND * 4);
  _Float16*  A16   = (_Float16*)alloc(ND * 2);
  _Float16*  Wt16  = (_Float16*)alloc((size_t)DD * DD * 2);
  float*     stats = (float*)alloc(2 * DD * 4);
  float*     a_s   = (float*)alloc((size_t)NN * HH * 4);
  float*     a_d   = (float*)alloc((size_t)NN * HH * 4);
  unsigned*  mx    = (unsigned*)alloc((size_t)NN * HH * 4);
  float*     den   = (float*)alloc((size_t)NN * HH * 4);
  float*     esc   = (float*)alloc((size_t)(EE + NN) * HH * 4);
  float*     gpool = (float*)alloc((size_t)GG * DD * 4);
  float*     gbuf  = (float*)alloc((size_t)GG * DD * 4);
  _Float16*  g16   = (_Float16*)alloc((size_t)GG * DD * 2);
  float*     xcat  = (float*)alloc((size_t)GG * 2 * HID * 4);
  _Float16*  x16   = (_Float16*)alloc((size_t)GG * 2 * HID * 2);
  float*     rh    = (float*)alloc((size_t)GG * HID * 4);

  run_branch(wl_x, wl_ei, wl_b, g1W1, g1b1, g1W2, g1b2, bn1g, bn1b,
             g2W1, g2b1, g2W2, g2b2, bn2g, bn2b, gatW, gatas, gatad, gatbias,
             fcW1, fcb1, fcbng, fcbnb, fcW2, fcb2,
             buf0, buf1, buf2, A16, Wt16, stats, a_s, a_d, mx, den, esc,
             gpool, gbuf, g16, xcat, /*col_off=*/0, stream);
  run_branch(mt_x, mt_ei, mt_b, g1W1, g1b1, g1W2, g1b2, bn1g, bn1b,
             g2W1, g2b1, g2W2, g2b2, bn2g, bn2b, gatW, gatas, gatad, gatbias,
             fcW1, fcb1, fcbng, fcbnb, fcW2, fcb2,
             buf0, buf1, buf2, A16, Wt16, stats, a_s, a_d, mx, den, esc,
             gpool, gbuf, g16, xcat, /*col_off=*/HID, stream);

  // ---- regressor: [64,256] -> relu(@W1+b1) -> [64,128] -> @W2+b2 -> [64] ----
  conv16(xcat, x16, GG * 2 * HID, stream);
  wtr(rW1, Wt16, 2 * HID, HID, 2 * HID, stream);
  gemm(x16, Wt16, rb1, rh, GG, 2 * HID, HID, HID, 0, 1, stream);
  reg_final_kernel<<<1, 64, 0, stream>>>(rh, rW2, rb2, (float*)d_out);
}
